// CausalSelfAttention_71021579206871
// MI455X (gfx1250) — compile-verified, matrix operand order checked by reading, not run on hardware
//
#include <hip/hip_runtime.h>

typedef _Float16 f16;
typedef __attribute__((ext_vector_type(16))) _Float16 v16h;
typedef __attribute__((ext_vector_type(8)))  _Float16 v8h;
typedef __attribute__((ext_vector_type(8)))  float    v8f;
typedef __attribute__((ext_vector_type(4)))  float    v4f;

#define BB 2
#define TT 2048
#define DD 1024
#define HH 16
#define HD 64

static __device__ __forceinline__ v8f wmma16(v16h a, v16h b, v8f c) {
    // D = A(16x32 f16) * B(32x16 f16) + C(16x16 f32)
    return __builtin_amdgcn_wmma_f32_16x16x32_f16(false, a, false, b, (short)0, c,
                                                  false, false);
}

// A/B fragment layout (16-bit, 16x32): lane L (row/col = L&15, half = L>>4)
// element i<8  -> K = 8*half + i ; element i>=8 -> K = 16 + 8*half + (i-8)
static __device__ __forceinline__ v16h load_frag_f16(const f16* __restrict__ base, int half) {
    v8h lo = *(const v8h*)(base + 8 * half);
    v8h hi = *(const v8h*)(base + 16 + 8 * half);
    v16h f;
#pragma unroll
    for (int i = 0; i < 8; ++i) { f[i] = lo[i]; f[i + 8] = hi[i]; }
    return f;
}

// ---------------------------------------------------------------------------
// Kernel 0: one-shot fp32 -> f16 conversion (x, W_qkv, W_out)
// ---------------------------------------------------------------------------
__global__ __launch_bounds__(256) void cvt_f16_kernel(const float* __restrict__ src,
                                                      f16* __restrict__ dst, int n8) {
    const int i = blockIdx.x * 256 + threadIdx.x;
    if (i >= n8) return;
    const v4f* s4 = (const v4f*)src + (size_t)i * 2;
    v4f x0 = s4[0], x1 = s4[1];
    v8h o;
#pragma unroll
    for (int j = 0; j < 4; ++j) { o[j] = (f16)x0[j]; o[j + 4] = (f16)x1[j]; }
    *((v8h*)dst + i) = o;
}

// ---------------------------------------------------------------------------
// Kernel 1: qkv = x @ W_qkv^T, one q/k/v section per blockIdx.z (uniform branch)
//   M = B*T = 4096, N = 1024 per section, K = 1024.  Wave tile 32x64 (2x4 WMMA).
//   Epilogue: q,k -> [bh, t, hd] f16 ; v -> transposed [bh, hd, t] f16 (b128 stores)
// ---------------------------------------------------------------------------
__global__ __launch_bounds__(128) void gemm_qkv_kernel(const f16* __restrict__ A,
                                                       const f16* __restrict__ Wf,
                                                       f16* __restrict__ qb,
                                                       f16* __restrict__ kb,
                                                       f16* __restrict__ vtb) {
    const int K    = DD;
    const int lane = threadIdx.x & 31, l16 = lane & 15, half = lane >> 4;
    const int w    = threadIdx.x >> 5;
    const int s    = blockIdx.z;                       // 0:q 1:k 2:v  (scalar!)
    const int m_base = blockIdx.y * 64  + (w >> 1) * 32;
    const int n_base = blockIdx.x * 128 + (w & 1) * 64;   // within 1024-wide section
    const f16* W = Wf + (size_t)s * DD * DD;              // section of W_qkv

    v8f acc[2][4] = {};
    const f16* a0p = A + (size_t)(m_base + l16) * K;
    const f16* a1p = A + (size_t)(m_base + 16 + l16) * K;
    const f16* bp[4];
#pragma unroll
    for (int j = 0; j < 4; ++j) bp[j] = W + (size_t)(n_base + j * 16 + l16) * K;

    for (int k0 = 0; k0 < K; k0 += 32) {
        v16h a0 = load_frag_f16(a0p + k0, half);
        v16h a1 = load_frag_f16(a1p + k0, half);
#pragma unroll
        for (int j = 0; j < 4; ++j) {
            v16h bf = load_frag_f16(bp[j] + k0, half);
            acc[0][j] = wmma16(a0, bf, acc[0][j]);
            acc[1][j] = wmma16(a1, bf, acc[1][j]);
        }
    }

    if (s == 2) {
        // V: store transposed [bh, hd, t]; 8 consecutive t per lane -> b128 store
#pragma unroll
        for (int ti = 0; ti < 2; ++ti) {
#pragma unroll
            for (int j = 0; j < 4; ++j) {
                const int nn = n_base + j * 16 + l16;   // 0..1023
                const int h  = nn >> 6, d = nn & (HD - 1);
                const int t_base = m_base + ti * 16 + 8 * half;
                const int b  = t_base >> 11;            // /TT
                const int t  = t_base & (TT - 1);
                v8h st;
#pragma unroll
                for (int r = 0; r < 8; ++r) st[r] = (f16)acc[ti][j][r];
                *(v8h*)(vtb + ((size_t)(b * HH + h) * HD + d) * TT + t) = st;
            }
        }
    } else {
        f16* dst = (s == 0) ? qb : kb;                  // uniform select
#pragma unroll
        for (int ti = 0; ti < 2; ++ti) {
#pragma unroll
            for (int j = 0; j < 4; ++j) {
                const int nn = n_base + j * 16 + l16;
                const int h  = nn >> 6, d = nn & (HD - 1);
#pragma unroll
                for (int r = 0; r < 8; ++r) {
                    const int m = m_base + ti * 16 + 8 * half + r;
                    const int b = m >> 11, t = m & (TT - 1);
                    dst[((size_t)(b * HH + h) * TT + t) * HD + d] = (f16)acc[ti][j][r];
                }
            }
        }
    }
}

// ---------------------------------------------------------------------------
// Kernel 2: flash attention, one wave per 16-query tile per (b,h).
// S^T = K*Q^T (lane holds one query column) -> online softmax in registers
// -> P feeds directly as B-fragment of O^T = V^T * P^T.
// ---------------------------------------------------------------------------
__global__ __launch_bounds__(128) void attn_kernel(const f16* __restrict__ Qb,
                                                   const f16* __restrict__ Kb,
                                                   const f16* __restrict__ Vtb,
                                                   f16* __restrict__ Ob) {
    const int lane = threadIdx.x & 31, l16 = lane & 15, half = lane >> 4;
    const int wid = blockIdx.x * 4 + (threadIdx.x >> 5);
    const int qt  = wid & (TT / 16 - 1);
    const int bh  = wid >> 7;
    const int h   = bh & (HH - 1);
    const int b   = bh >> 4;
    const int q0  = qt * 16;
    const int qq  = q0 + l16;

    const float slope = exp2f(-0.5f * (float)(h + 1));   // (2^(-8/H))^(h+1)
    const float scale = 0.125f;                          // 1/sqrt(64)

    const f16* Q  = Qb  + (size_t)bh * TT * HD;
    const f16* Kt = Kb  + (size_t)bh * TT * HD;
    const f16* Vt = Vtb + (size_t)bh * HD * TT;

    const f16* qrow = Q + (size_t)qq * HD;
    const v16h qf0 = load_frag_f16(qrow, half);
    const v16h qf1 = load_frag_f16(qrow + 32, half);

    float m_run = -1e30f, l_run = 0.0f;
    v8f o[4] = {};

    const int nchunk = (q0 + 47) >> 5;                   // keys 0 .. q0+15
    for (int c = 0; c < nchunk; ++c) {
        const int k0 = c * 32;
        v8f s0 = {}, s1 = {};
        const f16* kr0 = Kt + (size_t)(k0 + l16) * HD;
        const f16* kr1 = Kt + (size_t)(k0 + 16 + l16) * HD;
        v16h ka;
        ka = load_frag_f16(kr0, half);       s0 = wmma16(ka, qf0, s0);
        ka = load_frag_f16(kr0 + 32, half);  s0 = wmma16(ka, qf1, s0);
        ka = load_frag_f16(kr1, half);       s1 = wmma16(ka, qf0, s1);
        ka = load_frag_f16(kr1 + 32, half);  s1 = wmma16(ka, qf1, s1);

        float t0[8], t1[8];
        float mloc = -1e30f;
#pragma unroll
        for (int r = 0; r < 8; ++r) {
            const int kk0 = k0 + 8 * half + r;
            const int kk1 = kk0 + 16;
            float a0 = s0[r] * scale - slope * (float)(qq - kk0);
            float a1 = s1[r] * scale - slope * (float)(qq - kk1);
            if (kk0 > qq) a0 = -1e30f;
            if (kk1 > qq) a1 = -1e30f;
            t0[r] = a0; t1[r] = a1;
            mloc = fmaxf(mloc, fmaxf(a0, a1));
        }
        mloc = fmaxf(mloc, __shfl_xor(mloc, 16));

        const float mnew = fmaxf(m_run, mloc);
        const float corr = __expf(m_run - mnew);
        float lloc = 0.0f;
        v16h pf;                                          // P^T as next B-fragment
#pragma unroll
        for (int r = 0; r < 8; ++r) {
            const float p0 = __expf(t0[r] - mnew);
            const float p1 = __expf(t1[r] - mnew);
            lloc += p0 + p1;
            pf[r]     = (f16)p0;
            pf[r + 8] = (f16)p1;
        }
        lloc += __shfl_xor(lloc, 16);
        l_run = l_run * corr + lloc;
        m_run = mnew;
#pragma unroll
        for (int td = 0; td < 4; ++td)
#pragma unroll
            for (int r = 0; r < 8; ++r) o[td][r] *= corr;

        o[0] = wmma16(load_frag_f16(Vt + (size_t)(l16) * TT + k0, half),      pf, o[0]);
        o[1] = wmma16(load_frag_f16(Vt + (size_t)(16 + l16) * TT + k0, half), pf, o[1]);
        o[2] = wmma16(load_frag_f16(Vt + (size_t)(32 + l16) * TT + k0, half), pf, o[2]);
        o[3] = wmma16(load_frag_f16(Vt + (size_t)(48 + l16) * TT + k0, half), pf, o[3]);
    }

    const float linv = 1.0f / l_run;
    f16* orow = Ob + ((size_t)(b * TT + qq)) * DD + h * HD;
#pragma unroll
    for (int td = 0; td < 4; ++td) {
        v8h st;
#pragma unroll
        for (int r = 0; r < 8; ++r) st[r] = (f16)(o[td][r] * linv);
        *(v8h*)(orow + td * 16 + 8 * half) = st;
    }
}

// ---------------------------------------------------------------------------
// Kernel 3: out = attn_out(f16) @ W_out^T(f16) -> fp32 (M=4096, N=K=1024)
// Wave tile 32x64 (2x4 WMMA).
// ---------------------------------------------------------------------------
__global__ __launch_bounds__(128) void gemm_out_kernel(const f16* __restrict__ A,
                                                       const f16* __restrict__ Wf,
                                                       float* __restrict__ Out) {
    const int K    = DD;
    const int lane = threadIdx.x & 31, l16 = lane & 15, half = lane >> 4;
    const int w    = threadIdx.x >> 5;
    const int m_base = blockIdx.y * 64  + (w >> 1) * 32;
    const int n_base = blockIdx.x * 128 + (w & 1) * 64;

    v8f acc[2][4] = {};
    const f16* a0p = A + (size_t)(m_base + l16) * K;
    const f16* a1p = A + (size_t)(m_base + 16 + l16) * K;
    const f16* bp[4];
#pragma unroll
    for (int j = 0; j < 4; ++j) bp[j] = Wf + (size_t)(n_base + j * 16 + l16) * K;

    for (int k0 = 0; k0 < K; k0 += 32) {
        v16h a0 = load_frag_f16(a0p + k0, half);
        v16h a1 = load_frag_f16(a1p + k0, half);
#pragma unroll
        for (int j = 0; j < 4; ++j) {
            v16h bf = load_frag_f16(bp[j] + k0, half);
            acc[0][j] = wmma16(a0, bf, acc[0][j]);
            acc[1][j] = wmma16(a1, bf, acc[1][j]);
        }
    }

#pragma unroll
    for (int ti = 0; ti < 2; ++ti) {
#pragma unroll
        for (int j = 0; j < 4; ++j) {
            const int n = n_base + j * 16 + l16;
#pragma unroll
            for (int r = 0; r < 8; ++r) {
                const int m = m_base + ti * 16 + 8 * half + r;
                Out[(size_t)m * DD + n] = acc[ti][j][r];
            }
        }
    }
}

extern "C" void kernel_launch(void* const* d_in, const int* in_sizes, int n_in,
                              void* d_out, int out_size, void* d_ws, size_t ws_size,
                              hipStream_t stream) {
    const float* x    = (const float*)d_in[0];   // [B,T,D] fp32
    const float* Wqkv = (const float*)d_in[1];   // [3D,D]  fp32
    const float* Wout = (const float*)d_in[2];   // [D,D]   fp32
    (void)in_sizes; (void)n_in; (void)out_size; (void)ws_size;

    const size_t NX  = (size_t)BB * TT * DD;     // 4,194,304
    const size_t NWQ = (size_t)3 * DD * DD;      // 3,145,728
    const size_t NWO = (size_t)DD * DD;          // 1,048,576
    const size_t NE  = (size_t)BB * HH * TT * HD;

    f16* xf   = (f16*)d_ws;                      // x in f16
    f16* wqf  = xf + NX;                         // W_qkv in f16
    f16* wof  = wqf + NWQ;                       // W_out in f16
    f16* qb   = wof + NWO;
    f16* kb   = qb + NE;
    f16* vtb  = kb + NE;                         // V transposed [bh, hd, t]
    f16* ab   = vtb + NE;                        // attention out [B*T, D]

    dim3 blk(128);
    // 0) one-shot f16 conversion
    cvt_f16_kernel<<<dim3((int)(NX  / 8 / 256)), dim3(256), 0, stream>>>(x,    xf,  (int)(NX  / 8));
    cvt_f16_kernel<<<dim3((int)(NWQ / 8 / 256)), dim3(256), 0, stream>>>(Wqkv, wqf, (int)(NWQ / 8));
    cvt_f16_kernel<<<dim3((int)(NWO / 8 / 256)), dim3(256), 0, stream>>>(Wout, wof, (int)(NWO / 8));
    // 1) qkv projection: M=4096, N=1024 per section, z = q/k/v
    gemm_qkv_kernel<<<dim3(1024 / 128, 4096 / 64, 3), blk, 0, stream>>>(xf, wqf, qb, kb, vtb);
    // 2) flash attention: B*H*(T/16) = 4096 waves, 4 waves/block
    attn_kernel<<<dim3(1024), blk, 0, stream>>>(qb, kb, vtb, ab);
    // 3) output projection: M=4096, N=1024
    gemm_out_kernel<<<dim3(1024 / 128, 4096 / 64), blk, 0, stream>>>(ab, wof, (float*)d_out);
}
